// MultiHeadAttention_7164005449745
// MI455X (gfx1250) — matrix operand order repaired and checked by reading, not compiled, not run on hardware
//
#include <hip/hip_runtime.h>

typedef __attribute__((ext_vector_type(16))) _Float16 v16h;
typedef __attribute__((ext_vector_type(8)))  float    v8f;

#define EMB   1024
#define SEQ   2048
#define BATCH 2
#define HEADS 16
#define HD    64
#define BH    (BATCH*HEADS)      // 32
#define ROWS  (BATCH*SEQ)        // 4096

// ---------------------------------------------------------------------------
// CDNA5 async global->LDS (bypasses VGPRs, tracked by ASYNCcnt).
// LDS byte offset = low 32 bits of the flat shared-memory address
// (ISA 10.2).
// ---------------------------------------------------------------------------
__device__ __forceinline__ void async_load_b128(const void* gptr, void* lds_ptr)
{
    unsigned int lds_off = (unsigned int)(unsigned long long)lds_ptr;
    asm volatile("global_load_async_to_lds_b128 %0, %1, off"
                 :: "v"(lds_off), "v"(gptr) : "memory");
}

template<int N>
__device__ __forceinline__ void wait_asynccnt()
{
    asm volatile("s_wait_asynccnt %0" :: "i"(N) : "memory");
}

// ---------------------------------------------------------------------------
// VALU-only 16-lane reductions via DPP (replaces ds_bpermute shuffles).
// All controls permute within a 16-lane row, matching the WMMA C layout
// where the two 16-lane halves hold independent row groups.
// ---------------------------------------------------------------------------
template<int CTRL>
__device__ __forceinline__ float dpp_movf(float x)
{
    return __int_as_float(__builtin_amdgcn_update_dpp(
        __float_as_int(x), __float_as_int(x), CTRL, 0xf, 0xf, false));
}

__device__ __forceinline__ float red_max16(float x)
{
    x = fmaxf(x, dpp_movf<0xB1>(x));    // quad_perm(1,0,3,2): xor 1
    x = fmaxf(x, dpp_movf<0x4E>(x));    // quad_perm(2,3,0,1): xor 2
    x = fmaxf(x, dpp_movf<0x141>(x));   // row_half_mirror:   xor 7
    x = fmaxf(x, dpp_movf<0x140>(x));   // row_mirror:        xor 15
    return x;
}

__device__ __forceinline__ float red_sum16(float x)
{
    x += dpp_movf<0xB1>(x);
    x += dpp_movf<0x4E>(x);
    x += dpp_movf<0x141>(x);
    x += dpp_movf<0x140>(x);
    return x;
}

// ---------------------------------------------------------------------------
// f32 -> f16 conversion (vectorized 4-wide)
// ---------------------------------------------------------------------------
struct __align__(8) h4 { _Float16 x, y, z, w; };

__global__ void __launch_bounds__(256)
cvt_f32_to_f16(const float* __restrict__ in, _Float16* __restrict__ out, int n)
{
    int i = (blockIdx.x * 256 + threadIdx.x) * 4;
    if (i < n) {
        float4 v = *(const float4*)(in + i);
        h4 o{ (_Float16)v.x, (_Float16)v.y, (_Float16)v.z, (_Float16)v.w };
        *(h4*)(out + i) = o;
    }
}

// ---------------------------------------------------------------------------
// Tiled f16 GEMM, f32 accumulate:  C[M,N] = A[M,K] * B[N,K]^T + bias[N]
// MODE 0: f32 output, plain [M,N]             (final Wo projection)
// MODE 1: f16 output, scattered to [b,h,n,d]  (Q/K/V projections)
// Block: 256 thr (8 waves). Block tile 128x128, K-step 32, double-buffered
// LDS fed by async global->LDS. Wave grid 4x2; wave tile 32x64 = 8 WMMA/step.
// ---------------------------------------------------------------------------
#define TM  128
#define TN  128
#define TK  32
#define LDA 40   // padded LDS stride (halves): bank-spreading, 16B aligned
#define LDB 40

template<int MODE>
__global__ void __launch_bounds__(256)
gemm_f16_kernel(const _Float16* __restrict__ A, const _Float16* __restrict__ B,
                const float* __restrict__ bias,
                float* __restrict__ outF32, _Float16* __restrict__ outQKV,
                int M, int N, int K)
{
    __shared__ __align__(16) _Float16 As[2][TM * LDA];
    __shared__ __align__(16) _Float16 Bs[2][TN * LDB];

    const int tid  = threadIdx.x;
    const int lane = tid & 31;
    const int wid  = tid >> 5;
    const int wm   = wid & 3;     // 32-row slice within block tile
    const int wn   = wid >> 2;    // 64-col slice
    const int m15  = lane & 15;
    const int hi   = lane >> 4;   // half-wave select

    const int blockN0 = blockIdx.x * TN;
    const int blockM0 = blockIdx.y * TM;

    // staging: A tile 128x32 halves = 512 uint4 -> 2/thread, B same.
    const int ra0 = (tid)       >> 2, qa0 = (tid) & 3;
    const int ra1 = (tid + 256) >> 2, qa1 = (tid + 256) & 3;

    auto issue_tile = [&](int k0, int buf) {
        async_load_b128(A + (size_t)(blockM0 + ra0) * K + k0 + qa0 * 8,
                        &As[buf][ra0 * LDA + qa0 * 8]);
        async_load_b128(A + (size_t)(blockM0 + ra1) * K + k0 + qa1 * 8,
                        &As[buf][ra1 * LDA + qa1 * 8]);
        async_load_b128(B + (size_t)(blockN0 + ra0) * K + k0 + qa0 * 8,
                        &Bs[buf][ra0 * LDB + qa0 * 8]);
        async_load_b128(B + (size_t)(blockN0 + ra1) * K + k0 + qa1 * 8,
                        &Bs[buf][ra1 * LDB + qa1 * 8]);
    };

    v8f acc[2][4];
#pragma unroll
    for (int i = 0; i < 2; ++i)
#pragma unroll
        for (int j = 0; j < 4; ++j)
#pragma unroll
            for (int r = 0; r < 8; ++r) acc[i][j][r] = 0.0f;

    const int nk = K / TK;
    issue_tile(0, 0);

    for (int t = 0; t < nk; ++t) {
        const int buf = t & 1;
        if (t + 1 < nk) {
            issue_tile((t + 1) * TK, buf ^ 1);   // keep next tile in flight
            wait_asynccnt<4>();                  // tile t complete, t+1 pending
        } else {
            wait_asynccnt<0>();
        }
        __syncthreads();

        // A fragments (ISA 16-bit A layout: lane=M, K chunks interleave by hi)
        v16h af[2], bf[4];
#pragma unroll
        for (int i = 0; i < 2; ++i) {
            const _Float16* p = &As[buf][(wm * 32 + i * 16 + m15) * LDA + hi * 8];
            union { v16h v; uint4 q[2]; } u;
            u.q[0] = *(const uint4*)p;
            u.q[1] = *(const uint4*)(p + 16);
            af[i] = u.v;
        }
        // B fragments (lane=N, K chunks interleave by hi)
#pragma unroll
        for (int j = 0; j < 4; ++j) {
            const _Float16* p = &Bs[buf][(wn * 64 + j * 16 + m15) * LDB + hi * 8];
            union { v16h v; uint4 q[2]; } u;
            u.q[0] = *(const uint4*)p;
            u.q[1] = *(const uint4*)(p + 16);
            bf[j] = u.v;
        }
#pragma unroll
        for (int i = 0; i < 2; ++i)
#pragma unroll
            for (int j = 0; j < 4; ++j)
                acc[i][j] = __builtin_amdgcn_wmma_f32_16x16x32_f16(
                    false, af[i], false, bf[j], (short)0, acc[i][j], false, false);

        __syncthreads();   // all waves done with buf before it is overwritten
    }

    // Epilogue: C layout lane=col, VGPR=row (+8 for hi half-wave)
#pragma unroll
    for (int i = 0; i < 2; ++i)
#pragma unroll
        for (int j = 0; j < 4; ++j) {
            int col = blockN0 + wn * 64 + j * 16 + m15;
            float bv = bias[col];
#pragma unroll
            for (int r = 0; r < 8; ++r) {
                int row = blockM0 + wm * 32 + i * 16 + r + 8 * hi;
                float val = acc[i][j][r] + bv;
                if constexpr (MODE == 0) {
                    outF32[(size_t)row * N + col] = val;
                } else {
                    int b = row >> 11, n = row & 2047;   // row = b*SEQ + n
                    int h = col >> 6,  d = col & 63;     // col = h*HD + d
                    outQKV[(((size_t)(b * HEADS + h)) * SEQ + n) * HD + d] = (_Float16)val;
                }
            }
        }
}

// ---------------------------------------------------------------------------
// Flash attention: Q/K/V in [bh, n, d] f16. Block = 8 waves; wave owns 16
// queries; 64-key steps staged in LDS (K tile via async global->LDS, V tile
// transposed through VGPRs); online softmax with DPP reductions; O in f32.
// 16 WMMA per 64-key step between barriers.
// Output in concat layout [b, n, h*d] f16.
// ---------------------------------------------------------------------------
#define KT   64
#define KLS  72   // K-tile LDS stride (halves), 16B aligned, bank-spread
#define VLS  72   // Vt[d][key] stride (keys dimension now 64, padded)
#define PLS  64   // per-wave P tile row stride (16 rows x 64 keys)

__global__ void __launch_bounds__(256)
attn_kernel(const _Float16* __restrict__ Qp, const _Float16* __restrict__ Kp,
            const _Float16* __restrict__ Vp, _Float16* __restrict__ Outp)
{
    __shared__ __align__(16) _Float16 Kls[KT * KLS];       // [64 keys][64 d]
    __shared__ __align__(16) _Float16 Vt [HD * VLS];       // [64 d][64 keys]
    __shared__ __align__(16) _Float16 Pls[8 * 16 * PLS];   // per-wave P 16x64

    const int tid  = threadIdx.x;
    const int lane = tid & 31;
    const int wid  = tid >> 5;
    const int m15  = lane & 15;
    const int hi   = lane >> 4;

    const int bh = blockIdx.y;
    const int q0 = blockIdx.x * 128 + wid * 16;
    const _Float16* Qb = Qp + (size_t)bh * SEQ * HD;
    const _Float16* Kb = Kp + (size_t)bh * SEQ * HD;
    const _Float16* Vb = Vp + (size_t)bh * SEQ * HD;

    // Q A-fragments for d-chunks [0,32) and [32,64)
    v16h aq[2];
#pragma unroll
    for (int c = 0; c < 2; ++c) {
        const _Float16* p = Qb + (size_t)(q0 + m15) * HD + c * 32 + hi * 8;
        union { v16h v; uint4 q[2]; } u;
        u.q[0] = *(const uint4*)p;
        u.q[1] = *(const uint4*)(p + 16);
        aq[c] = u.v;
    }

    float mrow[8], lrow[8];
    v8f o[4];
#pragma unroll
    for (int i = 0; i < 8; ++i) { mrow[i] = -1e30f; lrow[i] = 0.0f; }
#pragma unroll
    for (int j = 0; j < 4; ++j)
#pragma unroll
        for (int i = 0; i < 8; ++i) o[j][i] = 0.0f;

    const float SCL = 0.03125f;  // 1/sqrt(EMB) = 1/32

    // staging: 64 rows x 64 halves = 512 uint4 jobs -> 2/thread
    const int r0 = (tid)       >> 3, s0q = (tid) & 7;
    const int r1 = (tid + 256) >> 3, s1q = (tid + 256) & 7;

    for (int k0 = 0; k0 < SEQ; k0 += KT) {
        __syncthreads();
        // K tile: direct async global->LDS (row-major)
        async_load_b128(Kb + (size_t)(k0 + r0) * HD + s0q * 8, Kls + r0 * KLS + s0q * 8);
        async_load_b128(Kb + (size_t)(k0 + r1) * HD + s1q * 8, Kls + r1 * KLS + s1q * 8);
        // V tile: through VGPRs, transposed into Vt[d][key]
        {
            union { uint4 u; _Float16 h[8]; } v0, v1;
            v0.u = *(const uint4*)(Vb + (size_t)(k0 + r0) * HD + s0q * 8);
            v1.u = *(const uint4*)(Vb + (size_t)(k0 + r1) * HD + s1q * 8);
#pragma unroll
            for (int e = 0; e < 8; ++e) Vt[(s0q * 8 + e) * VLS + r0] = v0.h[e];
#pragma unroll
            for (int e = 0; e < 8; ++e) Vt[(s1q * 8 + e) * VLS + r1] = v1.h[e];
        }
        wait_asynccnt<0>();
        __syncthreads();

        // S = Q * K^T : four 16-key tiles, K-dim = d = 64 -> 2 WMMA each
        v8f s[4];
#pragma unroll
        for (int kc = 0; kc < 4; ++kc) {
#pragma unroll
            for (int i = 0; i < 8; ++i) s[kc][i] = 0.0f;
#pragma unroll
            for (int dc = 0; dc < 2; ++dc) {
                const _Float16* p = Kls + (kc * 16 + m15) * KLS + dc * 32 + hi * 8;
                union { v16h v; uint4 q[2]; } u;
                u.q[0] = *(const uint4*)p;
                u.q[1] = *(const uint4*)(p + 16);
                s[kc] = __builtin_amdgcn_wmma_f32_16x16x32_f16(
                    false, aq[dc], false, u.v, (short)0, s[kc], false, false);
            }
        }

        // online softmax over 64 keys; VALU-only DPP reductions, then P->LDS
        _Float16* Pw = Pls + wid * (16 * PLS);
#pragma unroll
        for (int i = 0; i < 8; ++i) {
            float sv0 = s[0][i] * SCL, sv1 = s[1][i] * SCL;
            float sv2 = s[2][i] * SCL, sv3 = s[3][i] * SCL;
            float rm = fmaxf(fmaxf(sv0, sv1), fmaxf(sv2, sv3));
            rm = red_max16(rm);
            float mn   = fmaxf(mrow[i], rm);
            float corr = __expf(mrow[i] - mn);
            mrow[i] = mn;
            float e0 = __expf(sv0 - mn), e1 = __expf(sv1 - mn);
            float e2 = __expf(sv2 - mn), e3 = __expf(sv3 - mn);
            float rs = red_sum16((e0 + e1) + (e2 + e3));
            lrow[i] = lrow[i] * corr + rs;
#pragma unroll
            for (int j = 0; j < 4; ++j) o[j][i] *= corr;
            int row = i + 8 * hi;
            Pw[row * PLS + m15]      = (_Float16)e0;
            Pw[row * PLS + 16 + m15] = (_Float16)e1;
            Pw[row * PLS + 32 + m15] = (_Float16)e2;
            Pw[row * PLS + 48 + m15] = (_Float16)e3;
        }
        __syncthreads();

        // P A-fragments (two K-chunks of 32 keys)
        v16h ap[2];
#pragma unroll
        for (int c = 0; c < 2; ++c) {
            const _Float16* p = Pw + m15 * PLS + c * 32 + hi * 8;
            union { v16h v; uint4 q[2]; } u;
            u.q[0] = *(const uint4*)p;
            u.q[1] = *(const uint4*)(p + 16);
            ap[c] = u.v;
        }
        // O += P * V : 4 output d-tiles x 2 key-chunks = 8 WMMA
#pragma unroll
        for (int j = 0; j < 4; ++j) {
#pragma unroll
            for (int c = 0; c < 2; ++c) {
                const _Float16* p = Vt + (j * 16 + m15) * VLS + c * 32 + hi * 8;
                union { v16h v; uint4 q[2]; } u;
                u.q[0] = *(const uint4*)p;
                u.q[1] = *(const uint4*)(p + 16);
                o[j] = __builtin_amdgcn_wmma_f32_16x16x32_f16(
                    false, ap[c], false, u.v, (short)0, o[j], false, false);
            }
        }
    }

    // normalize + store in concat layout [b, n, h*HD + d] (f16)
    const int b = bh >> 4, h = bh & 15;
#pragma unroll
    for (int i = 0; i < 8; ++i) {
        float inv = 1.0f / lrow[i];
        int row = q0 + i + 8 * hi;
#pragma unroll
        for (int j = 0; j < 4; ++j) {
            int d = j * 16 + m15;
            Outp[((size_t)(b * SEQ + row)) * EMB + h * HD + d] = (_Float16)(o[j][i] * inv);
        }
    }
}

// ---------------------------------------------------------------------------
// Host launcher
// ---------------------------------------------------------------------------
extern "C" void kernel_launch(void* const* d_in, const int* in_sizes, int n_in,
                              void* d_out, int out_size, void* d_ws, size_t ws_size,
                              hipStream_t stream)
{
    const float* x  = (const float*)d_in[0];
    const float* Wq = (const float*)d_in[1];
    const float* bq = (const float*)d_in[2];
    const float* Wk = (const float*)d_in[3];
    const float* bk = (const float*)d_in[4];
    const float* Wv = (const float*)d_in[5];
    const float* bv = (const float*)d_in[6];
    const float* Wo = (const float*)d_in[7];
    const float* bo = (const float*)d_in[8];
    float* out = (float*)d_out;

    _Float16* ws = (_Float16*)d_ws;
    size_t off = 0;
    _Float16* xh  = ws + off; off += (size_t)ROWS * EMB;
    _Float16* Wqh = ws + off; off += (size_t)EMB * EMB;
    _Float16* Wkh = ws + off; off += (size_t)EMB * EMB;
    _Float16* Wvh = ws + off; off += (size_t)EMB * EMB;
    _Float16* Woh = ws + off; off += (size_t)EMB * EMB;
    _Float16* Qh  = ws + off; off += (size_t)BH * SEQ * HD;
    _Float16* Kh  = ws + off; off += (size_t)BH * SEQ * HD;
    _Float16* Vh  = ws + off; off += (size_t)BH * SEQ * HD;
    _Float16* Ah  = ws + off; off += (size_t)ROWS * EMB;

    // f32 -> f16 conversions
    cvt_f32_to_f16<<<(ROWS * EMB) / 1024, 256, 0, stream>>>(x,  xh,  ROWS * EMB);
    cvt_f32_to_f16<<<(EMB * EMB) / 1024, 256, 0, stream>>>(Wq, Wqh, EMB * EMB);
    cvt_f32_to_f16<<<(EMB * EMB) / 1024, 256, 0, stream>>>(Wk, Wkh, EMB * EMB);
    cvt_f32_to_f16<<<(EMB * EMB) / 1024, 256, 0, stream>>>(Wv, Wvh, EMB * EMB);
    cvt_f32_to_f16<<<(EMB * EMB) / 1024, 256, 0, stream>>>(Wo, Woh, EMB * EMB);

    // Q/K/V projections (f16 out, head-major layout)
    dim3 gg(EMB / TN, ROWS / TM);  // (8, 32)
    gemm_f16_kernel<1><<<gg, 256, 0, stream>>>(xh, Wqh, bq, nullptr, Qh, ROWS, EMB, EMB);
    gemm_f16_kernel<1><<<gg, 256, 0, stream>>>(xh, Wkh, bk, nullptr, Kh, ROWS, EMB, EMB);
    gemm_f16_kernel<1><<<gg, 256, 0, stream>>>(xh, Wvh, bv, nullptr, Vh, ROWS, EMB, EMB);

    // flash attention -> concat layout f16
    attn_kernel<<<dim3(SEQ / 128, BH), 256, 0, stream>>>(Qh, Kh, Vh, Ah);

    // output projection (f32 out)
    gemm_f16_kernel<0><<<gg, 256, 0, stream>>>(Ah, Woh, bo, out, nullptr, ROWS, EMB, EMB);
}
